// Model_4063039062442
// MI455X (gfx1250) — matrix operand order repaired and checked by reading, not compiled, not run on hardware
//
#include <hip/hip_runtime.h>
#include <hip/hip_bf16.h>
#include <math.h>

// ---------------------------------------------------------------------------
// Model dims (fixed by the reference)
// ---------------------------------------------------------------------------
#define ND   2048   // diseases
#define NPR  2048   // drugs
#define DD   128    // disease emb dim
#define DP   128    // drug emb dim
#define LAT  128
#define AT   32
#define BB   16384  // batch

typedef _Float16 half_t;
typedef __attribute__((ext_vector_type(16))) _Float16 v16h;
typedef __attribute__((ext_vector_type(8)))  _Float16 v8h;
typedef __attribute__((ext_vector_type(8)))  float    v8f;

union F16x16 { v16h v; v8h h[2]; };

// ---------------------------------------------------------------------------
// Utility kernels
// ---------------------------------------------------------------------------
__global__ void k_convert_f16(const float* __restrict__ in, half_t* __restrict__ out, size_t n) {
    for (size_t i = (size_t)blockIdx.x * blockDim.x + threadIdx.x; i < n;
         i += (size_t)gridDim.x * blockDim.x)
        out[i] = (half_t)in[i];
}

// out[C][R] = in[R][C]^T  (f32 -> f16)
__global__ void k_transpose_f16(const float* __restrict__ in, half_t* __restrict__ out,
                                int R, int C) {
    size_t n = (size_t)R * C;
    for (size_t i = (size_t)blockIdx.x * blockDim.x + threadIdx.x; i < n;
         i += (size_t)gridDim.x * blockDim.x) {
        size_t r = i / C, c = i % C;
        out[c * (size_t)R + r] = (half_t)in[i];
    }
}

__global__ void k_rowsum(const float* __restrict__ adj, float* __restrict__ deg, int N) {
    __shared__ float red[256];
    int i = blockIdx.x;
    float s = 0.f;
    for (int j = threadIdx.x; j < N; j += blockDim.x) s += adj[(size_t)i * N + j];
    red[threadIdx.x] = s; __syncthreads();
    for (int o = 128; o > 0; o >>= 1) {
        if ((int)threadIdx.x < o) red[threadIdx.x] += red[threadIdx.x + o];
        __syncthreads();
    }
    if (threadIdx.x == 0) deg[i] = red[0] + 1e-8f;
}

__global__ void k_bias_add(const float* __restrict__ a, const float* __restrict__ b,
                           float* __restrict__ out, int n) {
    int i = blockIdx.x * blockDim.x + threadIdx.x;
    if (i < n) out[i] = a[i] + b[i];
}

// out[M][32] = E[M][K] @ W[K][32]   (one wave per row)
__global__ void k_small_gemm32(const float* __restrict__ E, const float* __restrict__ W,
                               float* __restrict__ out, int K) {
    int row = blockIdx.x;
    int t = threadIdx.x;            // 0..31
    float acc = 0.f;
    const float* er = E + (size_t)row * K;
    for (int k = 0; k < K; ++k) acc += er[k] * W[k * AT + t];
    out[(size_t)row * AT + t] = acc;
}

// ---------------------------------------------------------------------------
// Fused attention score + row softmax -> f16 probability matrix
// alpha_ij = relu( sum_k relu(qs[i][k] + ko[j][k] + b1[k]) * W2[k] + b2 ) * adj[i][j]
// P[i][:]  = softmax(alpha[i][:])
// ---------------------------------------------------------------------------
__global__ void k_attention(const float* __restrict__ adjf,   // f32 adj (or null)
                            const half_t* __restrict__ adjh,  // f16 adj (or null)
                            const float* __restrict__ ko,     // [No][32]
                            const float* __restrict__ qs,     // [Ns][32]
                            const float* __restrict__ b1,     // [32]
                            const float* __restrict__ W2,     // [32]
                            const float* __restrict__ b2p,    // [1]
                            half_t* __restrict__ P,           // [Ns][No]
                            int No) {
    __shared__ float qb[AT];
    __shared__ float w2s[AT];
    __shared__ float arow[2048];
    __shared__ float red[256];

    const int i = blockIdx.x;
    const int tid = threadIdx.x;
    if (tid < AT) {
        qb[tid]  = qs[(size_t)i * AT + tid] + b1[tid];
        w2s[tid] = W2[tid];
    }
    __syncthreads();
    const float b2 = b2p[0];

    // pass 1: scores + max
    float mx = -3.4e38f;
    for (int j = tid; j < No; j += blockDim.x) {
        const float* kr = ko + (size_t)j * AT;
        float dot = 0.f;
#pragma unroll
        for (int k = 0; k < AT; ++k) {
            float v = kr[k] + qb[k];
            v = v > 0.f ? v : 0.f;
            dot += v * w2s[k];
        }
        float al = dot + b2;
        al = al > 0.f ? al : 0.f;
        float av = adjf ? adjf[(size_t)i * No + j] : (float)adjh[(size_t)i * No + j];
        al *= av;
        arow[j] = al;
        mx = fmaxf(mx, al);
    }
    red[tid] = mx; __syncthreads();
    for (int o = 128; o > 0; o >>= 1) {
        if (tid < o) red[tid] = fmaxf(red[tid], red[tid + o]);
        __syncthreads();
    }
    mx = red[0]; __syncthreads();

    // pass 2: exp + sum
    float s = 0.f;
    for (int j = tid; j < No; j += blockDim.x) {
        float e = __expf(arow[j] - mx);
        arow[j] = e;
        s += e;
    }
    red[tid] = s; __syncthreads();
    for (int o = 128; o > 0; o >>= 1) {
        if (tid < o) red[tid] += red[tid + o];
        __syncthreads();
    }
    const float inv = 1.f / red[0];

    // pass 3: write normalized probabilities (f16)
    for (int j = tid; j < No; j += blockDim.x)
        P[(size_t)i * No + j] = (half_t)(arow[j] * inv);
}

// ---------------------------------------------------------------------------
// WMMA GEMM, 32x32 macro-tile per wave (2x2 of 16x16 wmma sub-tiles):
//   C[M][N] = act( A[M][K](f16) @ BT[N][K](f16)^T + Cin + bias )
// Per K-step(32): 8 x b128 loads feed 4 x v_wmma_f32_16x16x32_f16
// (2 loads/wmma vs 4 for the naive tile -> halves L2 traffic).
// act: 0 = none, 1 = SELU.  Requires M%32==0, N%32==0, K%32==0.
// ---------------------------------------------------------------------------
__global__ void k_gemm_wmma(const half_t* __restrict__ A, const half_t* __restrict__ BT,
                            const float* __restrict__ Cin, const float* __restrict__ bias,
                            float* __restrict__ Cout, half_t* __restrict__ Couth,
                            int M, int N, int K, int act) {
    const int lane = threadIdx.x;     // 0..31 (wave32)
    const int n4   = lane & 15;
    const int hi   = lane >> 4;       // 0 or 1
    const int tileN = blockIdx.x * 32;
    const int tileM = blockIdx.y * 32;

    // A fragment rows (m = lane&15) for the two M sub-tiles
    const half_t* a0 = A  + (size_t)(tileM + n4) * K;
    const half_t* a1 = A  + (size_t)(tileM + 16 + n4) * K;
    // B fragment rows (n = lane&15) for the two N sub-tiles
    const half_t* b0 = BT + (size_t)(tileN + n4) * K;
    const half_t* b1 = BT + (size_t)(tileN + 16 + n4) * K;

    v8f c00 = {}, c01 = {}, c10 = {}, c11 = {};
    for (int k0 = 0; k0 < K; k0 += 32) {
        // prefetch the next 512B K-block of the streaming panels (speculative,
        // lowers to global_prefetch_b8; dropped silently at the tail)
        if ((k0 & 255) == 0 && k0 + 256 < K) {
            __builtin_prefetch(a0 + k0 + 256, 0, 0);
            __builtin_prefetch(a1 + k0 + 256, 0, 0);
            __builtin_prefetch(b0 + k0 + 256, 0, 0);
            __builtin_prefetch(b1 + k0 + 256, 0, 0);
        }
        F16x16 af0, af1, bf0, bf1;
        // A 16x32 f16 fragment: lanes 0-15 hold K 0-7 / 16-23, lanes 16-31 hold 8-15 / 24-31
        af0.h[0] = *(const v8h*)(a0 + k0 + 8 * hi);
        af0.h[1] = *(const v8h*)(a0 + k0 + 16 + 8 * hi);
        af1.h[0] = *(const v8h*)(a1 + k0 + 8 * hi);
        af1.h[1] = *(const v8h*)(a1 + k0 + 16 + 8 * hi);
        // B 32x16 f16 fragment: lanes 0-15 hold K 0-7 / 16-23, lanes 16-31 hold 8-15 / 24-31
        bf0.h[0] = *(const v8h*)(b0 + k0 + 8 * hi);
        bf0.h[1] = *(const v8h*)(b0 + k0 + 16 + 8 * hi);
        bf1.h[0] = *(const v8h*)(b1 + k0 + 8 * hi);
        bf1.h[1] = *(const v8h*)(b1 + k0 + 16 + 8 * hi);
        c00 = __builtin_amdgcn_wmma_f32_16x16x32_f16(false, af0.v, false, bf0.v,
                                                     (short)0, c00, false, false);
        c01 = __builtin_amdgcn_wmma_f32_16x16x32_f16(false, af0.v, false, bf1.v,
                                                     (short)0, c01, false, false);
        c10 = __builtin_amdgcn_wmma_f32_16x16x32_f16(false, af1.v, false, bf0.v,
                                                     (short)0, c10, false, false);
        c11 = __builtin_amdgcn_wmma_f32_16x16x32_f16(false, af1.v, false, bf1.v,
                                                     (short)0, c11, false, false);
    }

    auto epilogue = [&](const v8f& c, int rowBase, int colBase) {
#pragma unroll
        for (int r = 0; r < 8; ++r) {
            const int row = rowBase + r + 8 * hi;   // C VGPR r: M=r (lanes 0-15) / r+8 (16-31)
            const int col = colBase + n4;
            const size_t idx = (size_t)row * N + col;
            float v = c[r];
            if (Cin)  v += Cin[idx];
            if (bias) v += bias[col];
            if (act == 1) {
                const float kS = 1.0507009873554805f, kA = 1.6732632423543772f;
                v = (v > 0.f) ? kS * v : kS * kA * (__expf(v) - 1.f);
            }
            if (Cout)  Cout[idx]  = v;
            if (Couth) Couth[idx] = (half_t)v;
        }
    };
    epilogue(c00, tileM,      tileN);
    epilogue(c01, tileM,      tileN + 16);
    epilogue(c10, tileM + 16, tileN);
    epilogue(c11, tileM + 16, tileN + 16);
}

// out_h[i][c] = f16( X[i][c] / deg[i] )
__global__ void k_scale_div(const float* __restrict__ X, const float* __restrict__ deg,
                            half_t* __restrict__ out, int M, int N) {
    size_t n = (size_t)M * N;
    for (size_t i = (size_t)blockIdx.x * blockDim.x + threadIdx.x; i < n;
         i += (size_t)gridDim.x * blockDim.x)
        out[i] = (half_t)(X[i] / deg[i / N]);
}

// x[b][k] = f16( h_e[idx_d[b]][k] * h_p[idx_p[b]][k] )
__global__ void k_pair(const float* __restrict__ he, const float* __restrict__ hp,
                       const int* __restrict__ idd, const int* __restrict__ idp,
                       half_t* __restrict__ x, int B, int N) {
    size_t n = (size_t)B * N;
    for (size_t i = (size_t)blockIdx.x * blockDim.x + threadIdx.x; i < n;
         i += (size_t)gridDim.x * blockDim.x) {
        size_t b = i / N, k = i % N;
        x[i] = (half_t)(he[(size_t)idd[b] * N + k] * hp[(size_t)idp[b] * N + k]);
    }
}

// z = x2 @ Wpred + bpred; sigmoid -> out; BCE terms -> terms
__global__ void k_head(const half_t* __restrict__ x2, const float* __restrict__ Wpred,
                       const float* __restrict__ bpredp, const int* __restrict__ labels,
                       float* __restrict__ sig, float* __restrict__ terms, int B) {
    int b = blockIdx.x * blockDim.x + threadIdx.x;
    if (b >= B) return;
    const half_t* xr = x2 + (size_t)b * DD;
    float acc = 0.f;
#pragma unroll 8
    for (int k = 0; k < DD; ++k) acc += (float)xr[k] * Wpred[k];
    float z = acc + bpredp[0];
    float y = (float)labels[b];
    sig[b] = 1.f / (1.f + expf(-z));
    terms[b] = fmaxf(z, 0.f) - z * y + log1pf(expf(-fabsf(z)));
}

__global__ void k_zero1(float* p) { if (threadIdx.x == 0 && blockIdx.x == 0) p[0] = 0.f; }

__global__ void k_reduce_mean(const float* __restrict__ terms, float* __restrict__ out, int B) {
    __shared__ float red[256];
    float s = 0.f;
    for (int i = blockIdx.x * blockDim.x + threadIdx.x; i < B; i += gridDim.x * blockDim.x)
        s += terms[i];
    red[threadIdx.x] = s; __syncthreads();
    for (int o = 128; o > 0; o >>= 1) {
        if ((int)threadIdx.x < o) red[threadIdx.x] += red[threadIdx.x + o];
        __syncthreads();
    }
    if (threadIdx.x == 0) atomicAdd(out, red[0] / (float)B);
}

// ---------------------------------------------------------------------------
// Launcher
// ---------------------------------------------------------------------------
extern "C" void kernel_launch(void* const* d_in, const int* in_sizes, int n_in,
                              void* d_out, int out_size, void* d_ws, size_t ws_size,
                              hipStream_t stream) {
    (void)in_sizes; (void)n_in; (void)out_size; (void)ws_size;

    const float* e_p_adj   = (const float*)d_in[0];
    const float* e_e_adj   = (const float*)d_in[1];
    const float* p_p_adj   = (const float*)d_in[2];
    const int*   in_dis    = (const int*)d_in[3];
    const int*   in_drug   = (const int*)d_in[4];
    const int*   labels    = (const int*)d_in[5];
    const float* dis_emb   = (const float*)d_in[6];
    const float* drug_emb  = (const float*)d_in[7];
    const float* Wdg       = (const float*)d_in[8];
    const float* bdg_lin   = (const float*)d_in[9];
    const float* bdg       = (const float*)d_in[10];
    const float* Wpg       = (const float*)d_in[11];
    const float* bpg_lin   = (const float*)d_in[12];
    const float* bpg       = (const float*)d_in[13];
    const float* Wa1_d     = (const float*)d_in[14];
    const float* ba1_d     = (const float*)d_in[15];
    const float* Wa2_d     = (const float*)d_in[16];
    const float* ba2_d     = (const float*)d_in[17];
    const float* Wa1_p     = (const float*)d_in[18];
    const float* ba1_p     = (const float*)d_in[19];
    const float* Wa2_p     = (const float*)d_in[20];
    const float* ba2_p     = (const float*)d_in[21];
    const float* Wd2       = (const float*)d_in[22];
    const float* bd2       = (const float*)d_in[23];
    const float* Wp3       = (const float*)d_in[24];
    const float* bp3       = (const float*)d_in[25];
    const float* mlp_W1    = (const float*)d_in[26];
    const float* mlp_b1    = (const float*)d_in[27];
    const float* mlp_W2    = (const float*)d_in[28];
    const float* mlp_b2    = (const float*)d_in[29];
    const float* Wpred     = (const float*)d_in[30];
    const float* bpred     = (const float*)d_in[31];

    float* out = (float*)d_out;         // [0]=loss, [1..B]=sigmoid(z)

    // ---- workspace arena ----
    char* base = (char*)d_ws;
    size_t off = 0;
    auto arena = [&](size_t bytes) -> void* {
        void* p = (void*)(base + off);
        off += (bytes + 255) & ~(size_t)255;
        return p;
    };
    half_t* adjE_h   = (half_t*)arena((size_t)ND * ND * 2);
    half_t* adjP_h   = (half_t*)arena((size_t)NPR * NPR * 2);
    half_t* epT_h    = (half_t*)arena((size_t)NPR * ND * 2);   // e_p_adj^T (f16)
    half_t* ET_dis   = (half_t*)arena((size_t)DD * ND * 2);    // disease_emb^T
    half_t* ET_drug  = (half_t*)arena((size_t)DP * NPR * 2);   // drug_emb^T
    half_t* WdgT     = (half_t*)arena((size_t)LAT * DP * 2);
    half_t* WpgT     = (half_t*)arena((size_t)LAT * DD * 2);
    half_t* Wd2T     = (half_t*)arena((size_t)LAT * DD * 2);
    half_t* Wp3T     = (half_t*)arena((size_t)LAT * DP * 2);
    half_t* W1T      = (half_t*)arena((size_t)DD * LAT * 2);
    half_t* W2T      = (half_t*)arena((size_t)DD * DD * 2);
    float*  ko_d     = (float*)arena((size_t)NPR * AT * 4);
    float*  qs_d     = (float*)arena((size_t)ND * AT * 4);
    float*  ko_p     = (float*)arena((size_t)ND * AT * 4);
    float*  qs_p     = (float*)arena((size_t)NPR * AT * 4);
    float*  cb_e     = (float*)arena(LAT * 4);
    float*  cb_p     = (float*)arena(LAT * 4);
    float*  deg_d    = (float*)arena(ND * 4);
    float*  deg_p    = (float*)arena(NPR * 4);
    half_t* P_d      = (half_t*)arena((size_t)ND * NPR * 2);
    half_t* P_p      = (half_t*)arena((size_t)NPR * ND * 2);
    float*  h_e_att  = (float*)arena((size_t)ND * DP * 4);
    half_t* h_e_atth = (half_t*)arena((size_t)ND * DP * 2);
    float*  h_p_att  = (float*)arena((size_t)NPR * DD * 4);
    half_t* h_p_atth = (half_t*)arena((size_t)NPR * DD * 2);
    float*  Xe       = (float*)arena((size_t)ND * DD * 4);
    half_t* Xe_h     = (half_t*)arena((size_t)ND * DD * 2);
    float*  Xp       = (float*)arena((size_t)NPR * DP * 4);
    half_t* Xp_h     = (half_t*)arena((size_t)NPR * DP * 2);
    float*  acc_e    = (float*)arena((size_t)ND * LAT * 4);
    float*  acc_p    = (float*)arena((size_t)NPR * LAT * 4);
    float*  h_e      = (float*)arena((size_t)ND * LAT * 4);
    float*  h_p      = (float*)arena((size_t)NPR * LAT * 4);
    half_t* x_h      = (half_t*)arena((size_t)BB * LAT * 2);
    half_t* x1_h     = (half_t*)arena((size_t)BB * DD * 2);
    half_t* x2_h     = (half_t*)arena((size_t)BB * DD * 2);
    float*  terms    = (float*)arena((size_t)BB * 4);

    // ---- precompute: conversions / transposes / degrees / projections ----
    k_convert_f16<<<512, 256, 0, stream>>>(e_e_adj, adjE_h, (size_t)ND * ND);
    k_convert_f16<<<512, 256, 0, stream>>>(p_p_adj, adjP_h, (size_t)NPR * NPR);
    k_transpose_f16<<<512, 256, 0, stream>>>(e_p_adj, epT_h, ND, NPR);
    k_transpose_f16<<<256, 256, 0, stream>>>(dis_emb,  ET_dis,  ND, DD);
    k_transpose_f16<<<256, 256, 0, stream>>>(drug_emb, ET_drug, NPR, DP);
    k_transpose_f16<<<32, 256, 0, stream>>>(Wdg,    WdgT, DP,  LAT);
    k_transpose_f16<<<32, 256, 0, stream>>>(Wpg,    WpgT, DD,  LAT);
    k_transpose_f16<<<32, 256, 0, stream>>>(Wd2,    Wd2T, DD,  LAT);
    k_transpose_f16<<<32, 256, 0, stream>>>(Wp3,    Wp3T, DP,  LAT);
    k_transpose_f16<<<32, 256, 0, stream>>>(mlp_W1, W1T,  LAT, DD);
    k_transpose_f16<<<32, 256, 0, stream>>>(mlp_W2, W2T,  DD,  DD);
    k_rowsum<<<ND,  256, 0, stream>>>(e_e_adj, deg_d, ND);
    k_rowsum<<<NPR, 256, 0, stream>>>(p_p_adj, deg_p, NPR);
    // attention projections (split-concat trick): key = W1[:do], query = W1[do:]
    k_small_gemm32<<<NPR, 32, 0, stream>>>(drug_emb, Wa1_d,            ko_d, DP);
    k_small_gemm32<<<ND,  32, 0, stream>>>(dis_emb,  Wa1_d + DP * AT,  qs_d, DD);
    k_small_gemm32<<<ND,  32, 0, stream>>>(dis_emb,  Wa1_p,            ko_p, DD);
    k_small_gemm32<<<NPR, 32, 0, stream>>>(drug_emb, Wa1_p + DD * AT,  qs_p, DP);
    k_bias_add<<<1, LAT, 0, stream>>>(bdg_lin, bdg, cb_e, LAT);
    k_bias_add<<<1, LAT, 0, stream>>>(bpg_lin, bpg, cb_p, LAT);

    // ---- attention scores + softmax (f16 prob matrices) ----
    k_attention<<<ND, 256, 0, stream>>>(e_p_adj, nullptr, ko_d, qs_d, ba1_d,
                                        Wa2_d, ba2_d, P_d, NPR);
    k_attention<<<NPR, 256, 0, stream>>>(nullptr, epT_h, ko_p, qs_p, ba1_p,
                                         Wa2_p, ba2_p, P_p, ND);

    // ---- big WMMA GEMMs: softmax @ emb  and  adj @ emb ----
    {
        dim3 g(DP / 32, ND / 32);   // 4 x 64 blocks, 32x32 tile per wave
        k_gemm_wmma<<<g, 32, 0, stream>>>(P_d, ET_drug, nullptr, nullptr,
                                          h_e_att, h_e_atth, ND, DP, NPR, 0);
        k_gemm_wmma<<<g, 32, 0, stream>>>(P_p, ET_dis, nullptr, nullptr,
                                          h_p_att, h_p_atth, NPR, DD, ND, 0);
        k_gemm_wmma<<<g, 32, 0, stream>>>(adjE_h, ET_dis, nullptr, nullptr,
                                          Xe, nullptr, ND, DD, ND, 0);
        k_gemm_wmma<<<g, 32, 0, stream>>>(adjP_h, ET_drug, nullptr, nullptr,
                                          Xp, nullptr, NPR, DP, NPR, 0);
    }
    k_scale_div<<<256, 256, 0, stream>>>(Xe, deg_d, Xe_h, ND, DD);
    k_scale_div<<<256, 256, 0, stream>>>(Xp, deg_p, Xp_h, NPR, DP);

    // ---- GCN projections + SELU combine ----
    {
        dim3 g(LAT / 32, ND / 32);
        // acc_e = h_e_att @ Wdg + (bdg_lin + bdg)
        k_gemm_wmma<<<g, 32, 0, stream>>>(h_e_atth, WdgT, nullptr, cb_e,
                                          acc_e, nullptr, ND, LAT, DP, 0);
        // h_e = selu(acc_e + (Xe/deg) @ Wd2 + bd2)
        k_gemm_wmma<<<g, 32, 0, stream>>>(Xe_h, Wd2T, acc_e, bd2,
                                          h_e, nullptr, ND, LAT, DD, 1);
        k_gemm_wmma<<<g, 32, 0, stream>>>(h_p_atth, WpgT, nullptr, cb_p,
                                          acc_p, nullptr, NPR, LAT, DD, 0);
        k_gemm_wmma<<<g, 32, 0, stream>>>(Xp_h, Wp3T, acc_p, bp3,
                                          h_p, nullptr, NPR, LAT, DP, 1);
    }

    // ---- pair features + MLP ----
    k_pair<<<1024, 256, 0, stream>>>(h_e, h_p, in_dis, in_drug, x_h, BB, LAT);
    {
        dim3 g(DD / 32, BB / 32);   // 4 x 512
        k_gemm_wmma<<<g, 32, 0, stream>>>(x_h, W1T, nullptr, mlp_b1,
                                          nullptr, x1_h, BB, DD, LAT, 1);
        k_gemm_wmma<<<g, 32, 0, stream>>>(x1_h, W2T, nullptr, mlp_b2,
                                          nullptr, x2_h, BB, DD, DD, 1);
    }

    // ---- head: logits, sigmoid, BCE loss ----
    k_head<<<(BB + 255) / 256, 256, 0, stream>>>(x2_h, Wpred, bpred, labels,
                                                 out + 1, terms, BB);
    k_zero1<<<1, 1, 0, stream>>>(out);
    k_reduce_mean<<<64, 256, 0, stream>>>(terms, out, BB);
}
